// PV2BEV_69715909149218
// MI455X (gfx1250) — matrix operand order repaired and checked by reading, not compiled, not run on hardware
//
#include <hip/hip_runtime.h>

// ---------------------------------------------------------------------------
// BEV pooling as dense GEMM with an on-the-fly count matrix.
//   out[b,c,z,x] = sum_{hw} A[b,c,hw] * M[b,hw,zx]   (zx = z*128+x)
//   A = f16 copy of features, M = f16 count matrix (counts <= 60, exact).
// GEMM: v_wmma_f32_16x16x32_f16, K split 8-way for occupancy (2560 waves),
// partial tiles combined with f32 atomic adds into d_out (zeroed per call).
// ---------------------------------------------------------------------------

typedef __attribute__((ext_vector_type(16))) _Float16 v16h;
typedef __attribute__((ext_vector_type(8)))  _Float16 v8h;
typedef __attribute__((ext_vector_type(8)))  float    v8f;
typedef __attribute__((ext_vector_type(4)))  unsigned u32x4;

#define BS   4
#define CH   64
#define HW   2816   // H*W = 32*88 = K dimension (88 * 32)
#define NZ   60
#define BEVZ 10
#define BEVX 128
#define BEVY 128
#define ZX   1280   // BEVZ*BEVX = N dimension (80 * 16)
#define NTILES (ZX / 16)      // 80
#define KSPLIT 8
#define KCHUNK (HW / KSPLIT)  // 352 = 11 steps of 32

// Workspace layout (halves):
//   A  : BS*CH*HW   f16 features, A[b][c][hw]
//   MT : BS*ZX*HW   f16 counts, transposed: MT[b][zx][hw]
// total ~30.3 MB

__global__ void pv2bev_zero(u32x4* __restrict__ p, long n4) {
    long i = (long)blockIdx.x * blockDim.x + threadIdx.x;
    if (i < n4) {
        u32x4 z = {0u, 0u, 0u, 0u};
        p[i] = z;
    }
}

__global__ void pv2bev_cvt(const float* __restrict__ x, _Float16* __restrict__ a, int n) {
    int i = blockIdx.x * blockDim.x + threadIdx.x;
    if (i < n) a[i] = (_Float16)x[i];
}

// One thread owns (b, hw). All NZ points of that thread touch only column hw
// of MT[b][*][hw]  ->  non-atomic f16 read-modify-write is race-free.
__global__ void pv2bev_counts(const int* __restrict__ fr, _Float16* __restrict__ mt) {
    int t = blockIdx.x * blockDim.x + threadIdx.x;   // 0 .. BS*HW-1
    if (t >= BS * HW) return;
    int b  = t / HW;
    int hw = t % HW;
    const int* f = fr + (size_t)b * NZ * HW * 3 + (size_t)hw * 3;
    _Float16* col = mt + (size_t)b * ZX * HW + hw;
    #pragma unroll 4
    for (int nz = 0; nz < NZ; ++nz) {
        int xx = f[0];
        int yy = f[1];
        int zz = f[2];
        f += HW * 3;
        if ((unsigned)xx < BEVX && (unsigned)yy < BEVY && (unsigned)zz < BEVZ) {
            size_t off = (size_t)(zz * BEVX + xx) * HW;
            col[off] = col[off] + (_Float16)1.0f;
        }
    }
}

union Frag { v8h h[2]; v16h v; };

// One wave per (batch, N-tile, K-chunk); 4 accumulators cover all 64 channels.
// Double-buffered K loop; partials combined into d_out with f32 atomics.
__global__ __launch_bounds__(256) void pv2bev_gemm(const _Float16* __restrict__ A,
                                                   const _Float16* __restrict__ MT,
                                                   float* __restrict__ out) {
    const int lane = threadIdx.x & 31;
    const int wid  = (int)((blockIdx.x * blockDim.x + threadIdx.x) >> 5);
    if (wid >= BS * NTILES * KSPLIT) return;   // grid is exact; uniform guard
    const int b    = wid / (NTILES * KSPLIT);
    const int rem  = wid % (NTILES * KSPLIT);
    const int nt   = rem / KSPLIT;
    const int ks   = rem % KSPLIT;
    const int half = lane >> 4;
    const int row  = lane & 15;
    const int kbeg = ks * KCHUNK;

    // A fragment (16x32 f16): lane row = (mtile*16 + row); within-lane halves:
    //   elems 0..15 -> K = k0 + 16*half + j
    const _Float16* Ab = A + (size_t)b * CH * HW + (size_t)row * HW + 16 * half + kbeg;
    // B fragment (32x16 f16): lane column N = nt*16 + row, 16 contiguous K at
    // K = k0 + 16*half + j  (MT stored N-major so this is one 32B run).
    const _Float16* Bb = MT + (size_t)b * ZX * HW + (size_t)(nt * 16 + row) * HW
                            + 16 * half + kbeg;

    v8f acc0 = {}, acc1 = {}, acc2 = {}, acc3 = {};

    // prologue: load k-step 0
    Frag c0, c1, c2, c3, n0, n1, n2, n3;
    v16h bc, bn;
    bc      = *(const v16h*)(Bb);
    c0.h[0] = *(const v8h*)(Ab + 0 * 16 * HW);
    c0.h[1] = *(const v8h*)(Ab + 0 * 16 * HW + 8);
    c1.h[0] = *(const v8h*)(Ab + 1 * 16 * HW);
    c1.h[1] = *(const v8h*)(Ab + 1 * 16 * HW + 8);
    c2.h[0] = *(const v8h*)(Ab + 2 * 16 * HW);
    c2.h[1] = *(const v8h*)(Ab + 2 * 16 * HW + 8);
    c3.h[0] = *(const v8h*)(Ab + 3 * 16 * HW);
    c3.h[1] = *(const v8h*)(Ab + 3 * 16 * HW + 8);

    #pragma unroll
    for (int k0 = 32; k0 < KCHUNK; k0 += 32) {
        // prefetch next k-step while current WMMAs execute
        bn      = *(const v16h*)(Bb + k0);
        n0.h[0] = *(const v8h*)(Ab + 0 * 16 * HW + k0);
        n0.h[1] = *(const v8h*)(Ab + 0 * 16 * HW + k0 + 8);
        n1.h[0] = *(const v8h*)(Ab + 1 * 16 * HW + k0);
        n1.h[1] = *(const v8h*)(Ab + 1 * 16 * HW + k0 + 8);
        n2.h[0] = *(const v8h*)(Ab + 2 * 16 * HW + k0);
        n2.h[1] = *(const v8h*)(Ab + 2 * 16 * HW + k0 + 8);
        n3.h[0] = *(const v8h*)(Ab + 3 * 16 * HW + k0);
        n3.h[1] = *(const v8h*)(Ab + 3 * 16 * HW + k0 + 8);

        acc0 = __builtin_amdgcn_wmma_f32_16x16x32_f16(false, c0.v, false, bc,
                                                      (short)0, acc0, false, false);
        acc1 = __builtin_amdgcn_wmma_f32_16x16x32_f16(false, c1.v, false, bc,
                                                      (short)0, acc1, false, false);
        acc2 = __builtin_amdgcn_wmma_f32_16x16x32_f16(false, c2.v, false, bc,
                                                      (short)0, acc2, false, false);
        acc3 = __builtin_amdgcn_wmma_f32_16x16x32_f16(false, c3.v, false, bc,
                                                      (short)0, acc3, false, false);

        bc = bn; c0 = n0; c1 = n1; c2 = n2; c3 = n3;   // eliminated by unroll
    }

    acc0 = __builtin_amdgcn_wmma_f32_16x16x32_f16(false, c0.v, false, bc,
                                                  (short)0, acc0, false, false);
    acc1 = __builtin_amdgcn_wmma_f32_16x16x32_f16(false, c1.v, false, bc,
                                                  (short)0, acc1, false, false);
    acc2 = __builtin_amdgcn_wmma_f32_16x16x32_f16(false, c2.v, false, bc,
                                                  (short)0, acc2, false, false);
    acc3 = __builtin_amdgcn_wmma_f32_16x16x32_f16(false, c3.v, false, bc,
                                                  (short)0, acc3, false, false);

    // C/D layout: VGPR g -> row (g + 8*half), col = row(=lane&15).
    // Combine K-split partials with relaxed agent-scope f32 atomic adds.
    float* ob = out + (size_t)b * CH * ZX + (size_t)(nt * 16 + row);
    #pragma unroll
    for (int g = 0; g < 8; ++g) {
        int c = 8 * half + g;
        __hip_atomic_fetch_add(&ob[(size_t)(0 * 16 + c) * ZX], acc0[g],
                               __ATOMIC_RELAXED, __HIP_MEMORY_SCOPE_AGENT);
        __hip_atomic_fetch_add(&ob[(size_t)(1 * 16 + c) * ZX], acc1[g],
                               __ATOMIC_RELAXED, __HIP_MEMORY_SCOPE_AGENT);
        __hip_atomic_fetch_add(&ob[(size_t)(2 * 16 + c) * ZX], acc2[g],
                               __ATOMIC_RELAXED, __HIP_MEMORY_SCOPE_AGENT);
        __hip_atomic_fetch_add(&ob[(size_t)(3 * 16 + c) * ZX], acc3[g],
                               __ATOMIC_RELAXED, __HIP_MEMORY_SCOPE_AGENT);
    }
}

extern "C" void kernel_launch(void* const* d_in, const int* in_sizes, int n_in,
                              void* d_out, int out_size, void* d_ws, size_t ws_size,
                              hipStream_t stream) {
    (void)in_sizes; (void)n_in; (void)ws_size;
    const float* x  = (const float*)d_in[0];
    const int*   fr = (const int*)d_in[1];
    float*       out = (float*)d_out;

    _Float16* A  = (_Float16*)d_ws;
    _Float16* MT = A + (size_t)BS * CH * HW;

    // 1) zero count matrix (28.8 MB of f16) and d_out (atomic accumulation target)
    long n4m = (long)BS * ZX * HW / 8;
    pv2bev_zero<<<(unsigned)((n4m + 255) / 256), 256, 0, stream>>>((u32x4*)MT, n4m);
    long n4o = (long)out_size / 4;
    pv2bev_zero<<<(unsigned)((n4o + 255) / 256), 256, 0, stream>>>((u32x4*)out, n4o);

    // 2) features f32 -> f16
    int na = BS * CH * HW;
    pv2bev_cvt<<<(na + 255) / 256, 256, 0, stream>>>(x, A, na);

    // 3) build counts (race-free column ownership, no atomics)
    pv2bev_counts<<<(BS * HW + 255) / 256, 256, 0, stream>>>(fr, MT);

    // 4) WMMA GEMM: 2560 waves = 320 blocks of 8 waves
    pv2bev_gemm<<<(BS * NTILES * KSPLIT * 32 + 255) / 256, 256, 0, stream>>>(A, MT, out);
}
